// BiMambaWrapper_39359080301031
// MI455X (gfx1250) — matrix operand order repaired and checked by reading, not compiled, not run on hardware
//
#include <hip/hip_runtime.h>
#include <hip/hip_bf16.h>
#include <math.h>

// ---------------- problem constants ----------------
#define DM   1024           // d_model
#define DI   2048           // d_inner
#define DS   16             // d_state
#define DR   64             // dt_rank
#define DC   4              // d_conv
#define BB   2
#define LL   4096
#define MM   (BB*LL)        // 8192 rows
#define XZW  (2*DI)         // 4096  (xz row width)
#define XDW  (DR+2*DS)      // 96    (x_dbl row width)
#define XDP  128            // padded x_dbl width for GEMM tiling

// GEMM macro-tile
#define BM 128
#define BN 64
#define BK 32

typedef _Float16 half8  __attribute__((ext_vector_type(8)));
typedef _Float16 half16 __attribute__((ext_vector_type(16)));
typedef float    float8 __attribute__((ext_vector_type(8)));

// ---------------- CDNA5 async copy global -> LDS (16B per lane) ----------------
__device__ __forceinline__ void async_load16(const void* g, void* l) {
    unsigned loff = (unsigned)(size_t)l;              // low 32 bits = LDS offset
    asm volatile("global_load_async_to_lds_b128 %0, %1, off"
                 :: "v"(loff), "v"(g) : "memory");
}
__device__ __forceinline__ void wait_async_le3() {
    asm volatile("s_wait_asynccnt 0x3" ::: "memory");
}
__device__ __forceinline__ void wait_async_0() {
    asm volatile("s_wait_asynccnt 0x0" ::: "memory");
}

// ---------------- f32 -> f16 convert ----------------
__global__ __launch_bounds__(256) void k_cvt16(const float* __restrict__ s,
                                               _Float16* __restrict__ d, long n) {
    long i = (long)blockIdx.x * 256 + threadIdx.x;
    if (i < n) d[i] = (_Float16)s[i];
}

// convert (rows x K) f32 -> f16 into a (rowsPad x K) buffer, zero-filling pad rows
__global__ __launch_bounds__(256) void k_cvt16_pad(const float* __restrict__ s,
                                                   _Float16* __restrict__ d,
                                                   int rows, int rowsPad, int K) {
    long i = (long)blockIdx.x * 256 + threadIdx.x;
    long n = (long)rowsPad * K;
    if (i >= n) return;
    int r = (int)(i / K);
    d[i] = (r < rows) ? (_Float16)s[i] : (_Float16)0.f;
}

// ---------------- tiled WMMA GEMM:  C(MxN) = A(MxK) * B(NxK)^T ----------------
// A,B f16 row-major; C f32 row-major with row stride Nstore (store cols < Nstore).
// Double-buffered async-to-LDS staging; each wave computes a 32x32 (2x2 WMMA) tile.
// flip: output row (b,t) -> (b, L-1-t).   accum: C += instead of C =.
__device__ __forceinline__ void issue_stage(
    const _Float16* __restrict__ A, const _Float16* __restrict__ Bw, int Kdim,
    _Float16* sA, _Float16* sB, int m0, int n0, int k0, int tid)
{
    // A tile: BM x BK halves = 512 chunks of 8 halves (16B); 2 chunks/thread
#pragma unroll
    for (int c = tid; c < BM * BK / 8; c += 256) {
        int row = c >> 2, kc = (c & 3) * 8;
        async_load16(A + (size_t)(m0 + row) * Kdim + k0 + kc, sA + row * BK + kc);
    }
    // B tile: BN x BK halves = 256 chunks; 1 chunk/thread
    {
        int row = tid >> 2, kc = (tid & 3) * 8;
        async_load16(Bw + (size_t)(n0 + row) * Kdim + k0 + kc, sB + row * BK + kc);
    }
}

__global__ __launch_bounds__(256) void k_wmma_gemm_tiled(
    const _Float16* __restrict__ A, const _Float16* __restrict__ Bw,
    float* __restrict__ C, int Mdim, int Nstore, int Kdim, int accum, int flip)
{
    __shared__ __align__(128) _Float16 sA[2][BM * BK];
    __shared__ __align__(128) _Float16 sB[2][BN * BK];

    const int tid   = threadIdx.x;
    const int lane  = tid & 31;
    const int wave  = tid >> 5;
    const int waveM = wave & 3;                 // 4 waves along M
    const int waveN = wave >> 2;                // 2 waves along N
    const int m0 = blockIdx.y * BM;
    const int n0 = blockIdx.x * BN;

    float8 zero8 = {0.f, 0.f, 0.f, 0.f, 0.f, 0.f, 0.f, 0.f};
    float8 acc[2][2];
#pragma unroll
    for (int i = 0; i < 2; ++i)
#pragma unroll
        for (int j = 0; j < 2; ++j) acc[i][j] = zero8;

    // per-lane fragment coordinates (within the 32-wide wave tiles)
    const int frow  = lane & 15;                // A row / B col within 16
    const int akoff = (lane >> 4) * 16;         // A: lanes>=16 take K 16..31
    const int bkoff = (lane >> 4) * 16;         // B: lanes>=16 take K 16..31

    const int NK = Kdim / BK;
    issue_stage(A, Bw, Kdim, sA[0], sB[0], m0, n0, 0, tid);

    for (int ks = 0; ks < NK; ++ks) {
        const int cur = ks & 1;
        if (ks + 1 < NK) {
            issue_stage(A, Bw, Kdim, sA[cur ^ 1], sB[cur ^ 1], m0, n0, (ks + 1) * BK, tid);
            wait_async_le3();                   // previous stage complete
        } else {
            wait_async_0();
        }
        __syncthreads();                        // staged tiles visible to all waves

        half16 af[2], bf[2];
#pragma unroll
        for (int mt = 0; mt < 2; ++mt) {        // A fragments (ds_load_b128 x2 each)
            const _Float16* base = &sA[cur][(waveM * 32 + mt * 16 + frow) * BK];
            half8 lo = *(const half8*)(base + akoff);
            half8 hi = *(const half8*)(base + akoff + 8);
#pragma unroll
            for (int i = 0; i < 8; ++i) { af[mt][i] = lo[i]; af[mt][i + 8] = hi[i]; }
        }
#pragma unroll
        for (int nt = 0; nt < 2; ++nt) {        // B fragments (32B contiguous)
            bf[nt] = *(const half16*)(&sB[cur][(waveN * 32 + nt * 16 + frow) * BK + bkoff]);
        }
#pragma unroll
        for (int mt = 0; mt < 2; ++mt)
#pragma unroll
            for (int nt = 0; nt < 2; ++nt)
                acc[mt][nt] = __builtin_amdgcn_wmma_f32_16x16x32_f16(
                    false, af[mt], false, bf[nt], (short)0, acc[mt][nt], false, false);

        __syncthreads();                        // done reading before next overwrite
    }

    // store: per 16x16 tile, lane n = (lane&15); VGPR r -> M = (lane>>4)*8 + r
    const int mb = (lane >> 4) * 8;
#pragma unroll
    for (int mt = 0; mt < 2; ++mt) {
#pragma unroll
        for (int nt = 0; nt < 2; ++nt) {
            int cn = n0 + waveN * 32 + nt * 16 + (lane & 15);
            if (cn >= Nstore) continue;         // padded-N guard (store only)
#pragma unroll
            for (int r = 0; r < 8; ++r) {
                int m = m0 + waveM * 32 + mt * 16 + mb + r;
                if (flip) { int b = m / LL, t = m % LL; m = b * LL + (LL - 1 - t); }
                size_t ci = (size_t)m * Nstore + cn;
                if (accum) C[ci] += acc[mt][nt][r]; else C[ci] = acc[mt][nt][r];
            }
        }
    }
}

// ---------------- depthwise causal conv (k=4) + bias + SiLU ----------------
__global__ __launch_bounds__(256) void k_conv_silu(
    const float* __restrict__ xz, const float* __restrict__ cw,
    const float* __restrict__ cb, float* __restrict__ xc32,
    _Float16* __restrict__ xc16, int rev)
{
    int idx = blockIdx.x * 256 + threadIdx.x;     // < MM*DI
    int d = idx & (DI - 1);
    int m = idx >> 11;                            // /DI
    int b = m >> 12;                              // /LL
    int t = m & (LL - 1);
    float acc = cb[d];
#pragma unroll
    for (int j = 0; j < DC; ++j) {
        int tp = t - (DC - 1) + j;
        if (tp >= 0) {
            int ts = rev ? (LL - 1 - tp) : tp;
            acc += cw[d * DC + j] * xz[(size_t)(b * LL + ts) * XZW + d];
        }
    }
    float sv = acc / (1.f + expf(-acc));          // silu
    xc32[idx] = sv;
    xc16[idx] = (_Float16)sv;
}

// ---------------- extract dt-rank slice of x_dbl as f16 ----------------
__global__ __launch_bounds__(256) void k_split_dt(
    const float* __restrict__ xdbl, _Float16* __restrict__ dst)
{
    int i = blockIdx.x * 256 + threadIdx.x;       // < MM*DR
    int m = i >> 6, r = i & (DR - 1);
    dst[i] = (_Float16)xdbl[(size_t)m * XDW + r];
}

// ---------------- dt = softplus(dt_pre + b_dt), in place ----------------
__global__ __launch_bounds__(256) void k_softplus(
    float* __restrict__ dt, const float* __restrict__ b_dt)
{
    int i = blockIdx.x * 256 + threadIdx.x;       // < MM*DI
    int d = i & (DI - 1);
    float x = dt[i] + b_dt[d];
    dt[i] = (x > 20.f) ? x : log1pf(expf(x));
}

// ---------------- selective scan: one thread per (b, d) channel ----------------
__global__ __launch_bounds__(256) void k_scan(
    const float* __restrict__ dt, const float* __restrict__ u,
    const float* __restrict__ xdbl, const float* __restrict__ A_log,
    float* __restrict__ y)
{
    int tid = blockIdx.x * 256 + threadIdx.x;     // < BB*DI
    int d = tid & (DI - 1);
    int b = tid >> 11;
    float Av[DS], st[DS];
#pragma unroll
    for (int s = 0; s < DS; ++s) { Av[s] = -expf(A_log[d * DS + s]); st[s] = 0.f; }

    for (int t = 0; t < LL; ++t) {
        size_t m = (size_t)b * LL + t;
        float dtv = dt[m * DI + d];
        float uv  = u[m * DI + d];
        const float* Bp = xdbl + m * XDW + DR;
        const float* Cp = Bp + DS;
        float yv = 0.f;
#pragma unroll
        for (int s = 0; s < DS; ++s) {
            float dA = expf(dtv * Av[s]);
            st[s] = dA * st[s] + dtv * Bp[s] * uv;
            yv += st[s] * Cp[s];
        }
        y[m * DI + d] = yv;
    }
}

// ---------------- gate: yg = (y + u*D) * silu(z) ----------------
__global__ __launch_bounds__(256) void k_gate(
    const float* __restrict__ y, const float* __restrict__ u,
    const float* __restrict__ xz, const float* __restrict__ Dv,
    _Float16* __restrict__ yg16, int rev)
{
    int idx = blockIdx.x * 256 + threadIdx.x;     // < MM*DI
    int d = idx & (DI - 1);
    int m = idx >> 11;
    int b = m >> 12;
    int t = m & (LL - 1);
    int zm = rev ? (b * LL + (LL - 1 - t)) : m;
    float z = xz[(size_t)zm * XZW + DI + d];
    float g = z / (1.f + expf(-z));
    yg16[idx] = (_Float16)((y[idx] + u[idx] * Dv[d]) * g);
}

// ---------------- host orchestration ----------------
extern "C" void kernel_launch(void* const* d_in, const int* in_sizes, int n_in,
                              void* d_out, int out_size, void* d_ws, size_t ws_size,
                              hipStream_t stream) {
    const float* h      = (const float*)d_in[0];
    const float* W_in   = (const float*)d_in[1];
    const float* W_out  = (const float*)d_in[2];
    const float* dirp[2][7];
    for (int q = 0; q < 7; ++q) { dirp[0][q] = (const float*)d_in[3 + q];
                                  dirp[1][q] = (const float*)d_in[10 + q]; }

    char* p = (char*)d_ws;
    auto alloc = [&](size_t bytes) -> void* {
        void* r = (void*)p; p += (bytes + 255) & ~(size_t)255; return r;
    };
    _Float16* h16     = (_Float16*)alloc((size_t)MM * DM * 2);
    _Float16* Win16   = (_Float16*)alloc((size_t)XZW * DM * 2);
    _Float16* Wout16  = (_Float16*)alloc((size_t)DM * DI * 2);
    _Float16* Wx16    = (_Float16*)alloc((size_t)XDP * DI * 2);   // padded 96->128 rows
    _Float16* Wdt16   = (_Float16*)alloc((size_t)DI * DR * 2);
    float*    xz      = (float*)   alloc((size_t)MM * XZW * 4);
    float*    xc32    = (float*)   alloc((size_t)MM * DI * 4);
    _Float16* xc16    = (_Float16*)alloc((size_t)MM * DI * 2);
    float*    xdbl    = (float*)   alloc((size_t)MM * XDW * 4);
    _Float16* xdbl16  = (_Float16*)alloc((size_t)MM * DR * 2);
    float*    dtbuf   = (float*)   alloc((size_t)MM * DI * 4);
    float*    ybuf    = (float*)   alloc((size_t)MM * DI * 4);
    _Float16* yg16    = (_Float16*)alloc((size_t)MM * DI * 2);
    (void)ws_size; (void)in_sizes; (void)n_in; (void)out_size;

    auto cvt = [&](const float* s, _Float16* d, long n) {
        k_cvt16<<<dim3((unsigned)((n + 255) / 256)), 256, 0, stream>>>(s, d, n);
    };
    // Nld = padded N covered by tiles; Nstore = logical N (row stride & bound)
    auto gemm = [&](const _Float16* A, const _Float16* Bw, float* C,
                    int Md, int Nld, int Nstore, int Kd, int accum, int flip) {
        dim3 grid(Nld / BN, Md / BM);
        k_wmma_gemm_tiled<<<grid, 256, 0, stream>>>(A, Bw, C, Md, Nstore, Kd, accum, flip);
    };

    // shared precompute (reverse input GEMM is just a flip of the forward one)
    cvt(h,     h16,    (long)MM * DM);
    cvt(W_in,  Win16,  (long)XZW * DM);
    cvt(W_out, Wout16, (long)DM * DI);
    gemm(h16, Win16, xz, MM, XZW, XZW, DM, 0, 0);        // xz = h @ W_in^T

    const unsigned gMD = (unsigned)((size_t)MM * DI / 256);
    for (int dir = 0; dir < 2; ++dir) {
        const float *cw = dirp[dir][0], *cb = dirp[dir][1], *Wx = dirp[dir][2],
                    *Wdt = dirp[dir][3], *bdt = dirp[dir][4],
                    *Alog = dirp[dir][5], *Dv = dirp[dir][6];
        k_cvt16_pad<<<(unsigned)(((size_t)XDP * DI + 255) / 256), 256, 0, stream>>>(
            Wx, Wx16, XDW, XDP, DI);
        cvt(Wdt, Wdt16, (long)DI * DR);

        k_conv_silu<<<gMD, 256, 0, stream>>>(xz, cw, cb, xc32, xc16, dir);
        gemm(xc16, Wx16, xdbl, MM, XDP, XDW, DI, 0, 0);  // x_dbl = x @ W_x^T (N padded)
        k_split_dt<<<(unsigned)((size_t)MM * DR / 256), 256, 0, stream>>>(xdbl, xdbl16);
        gemm(xdbl16, Wdt16, dtbuf, MM, DI, DI, DR, 0, 0);// dt_pre
        k_softplus<<<gMD, 256, 0, stream>>>(dtbuf, bdt);
        k_scan<<<(unsigned)(BB * DI / 256), 256, 0, stream>>>(dtbuf, xc32, xdbl, Alog, ybuf);
        k_gate<<<gMD, 256, 0, stream>>>(ybuf, xc32, xz, Dv, yg16, dir);
        // out = fwd ; out += flip(rev)
        gemm(yg16, Wout16, (float*)d_out, MM, DM, DM, DI, /*accum=*/dir, /*flip=*/dir);
    }
}
